// STM_56581899157855
// MI455X (gfx1250) — matrix operand order repaired and checked by reading, not compiled, not run on hardware
//
#include <hip/hip_runtime.h>

#define NQ   4096
#define NM   40960
#define CK   128
#define CV   257
#define TOPK 32
#define TAUINV (1.0f/0.07f)
#define NWAVE 8
#define MPW  (NM/NWAVE)     // 5120 memory rows per wave
#define MTILES (MPW/16)     // 320 tiles of 16 rows (even -> unroll by 2)
#define NEG_BIG (-3.0e38f)

typedef __attribute__((ext_vector_type(16))) _Float16 v16h;
typedef __attribute__((ext_vector_type(8)))  float    v8f;
typedef unsigned int u32x4 __attribute__((ext_vector_type(4)));

union Frag { v16h h; u32x4 u[2]; };

// ---------------------------------------------------------------------------
// Phase 1: L2-normalize along channel axis (stride-N axis) and emit row-major
// f16 [n][C] ready for WMMA fragment loads.
// src element (c, n) at c*N + n  ->  dst (n, c) at n*C + c  (f16)
// ---------------------------------------------------------------------------
template <int C>
__global__ __launch_bounds__(256) void norm_to_f16(const float* __restrict__ src,
                                                   _Float16* __restrict__ dst,
                                                   int N) {
    int n = blockIdx.x * blockDim.x + threadIdx.x;
    if (n >= N) return;
    float x[C];
    float ss = 0.0f;
#pragma unroll
    for (int c = 0; c < C; ++c) {              // coalesced across lanes
        x[c] = src[(size_t)c * N + n];
        ss += x[c] * x[c];
    }
    float inv = 1.0f / fmaxf(sqrtf(ss), 1e-12f);
    unsigned int* orow = (unsigned int*)(dst + (size_t)n * C);
#pragma unroll
    for (int c = 0; c < C; c += 2) {
        _Float16 h0 = (_Float16)(x[c] * inv);
        _Float16 h1 = (_Float16)(x[c + 1] * inv);
        unsigned int u0 = __builtin_bit_cast(unsigned short, h0);
        unsigned int u1 = __builtin_bit_cast(unsigned short, h1);
        orow[c >> 1] = (u1 << 16) | u0;
    }
}

// ---------------------------------------------------------------------------
// Phase 2: fused  logits = Kn(M=mem rows) x Qn(N=queries)  +  per-lane top-32
//          + merge + softmax + top-k V gather.
// One workgroup (8 waves) per 16-query tile; each wave scans 5120 mem rows.
// A-fragment loads are double-buffered in registers (tile t+1 issued before
// tile t's WMMAs) so the s_wait_loadcnt is covered by 4 WMMAs + top-k VALU.
// ---------------------------------------------------------------------------
__global__ __launch_bounds__(256) void attn_topk_kernel(
        const _Float16* __restrict__ Qh,   // [NQ][CK] f16, L2-normalized
        const _Float16* __restrict__ Kh,   // [NM][CK] f16, L2-normalized
        const float*    __restrict__ V,    // [CV][NM] f32
        float*          __restrict__ out)  // [CV][NQ] f32
{
    __shared__ float cvs[16][256];   // candidate scores per query (8 waves * 32)
    __shared__ int   cis[16][256];   // candidate memory indices
    __shared__ float selv[16][TOPK]; // selected top-32 scores (descending)
    __shared__ int   seli[16][TOPK]; // selected top-32 indices
    __shared__ float attnw[16][TOPK];// softmax weights

    const int t    = threadIdx.x;
    const int lane = t & 31;
    const int wv   = t >> 5;         // wave 0..7
    const int lmod = lane & 15;
    const int lhi  = lane >> 4;      // 0/1 lane half
    const int qtile = blockIdx.x * 16;

    // --- loop-invariant B fragments (query matrix, 32x16 per K-chunk) ---
    // B layout: lanes 0-15: K 0-7 & 16-23; lanes 16-31: K 8-15 & 24-31 (matches A)
    Frag bq[4];
    {
        const _Float16* qrow = Qh + (size_t)(qtile + lmod) * CK + (lhi ? 8 : 0);
#pragma unroll
        for (int kc = 0; kc < 4; ++kc)
        {
            bq[kc].u[0] = *(const u32x4*)(qrow + kc * 32);
            bq[kc].u[1] = *(const u32x4*)(qrow + kc * 32 + 16);
        }
    }

    // --- per-lane sorted (ascending) top-32 reservoir in registers ---
    float tv[TOPK];
    int   ti[TOPK];
#pragma unroll
    for (int i = 0; i < TOPK; ++i) { tv[i] = NEG_BIG; ti[i] = 0; }

    const int mbase = wv * MPW;
    const _Float16* kbase = Kh + (size_t)(mbase + lmod) * CK + (lhi ? 8 : 0);

    // Load one 16-row tile's A fragments (4 K-chunks, 2 x b128 each).
    // A layout (16x32 f16): lanes 0-15: K 0-7 & 16-23; lanes 16-31: K 8-15 & 24-31
    auto loadTile = [&](Frag (&a)[4], int tile) {
        const _Float16* p = kbase + (size_t)tile * (16 * CK);
#pragma unroll
        for (int kc = 0; kc < 4; ++kc) {
            a[kc].u[0] = *(const u32x4*)(p + kc * 32);
            a[kc].u[1] = *(const u32x4*)(p + kc * 32 + 16);
        }
    };

    // 4 WMMAs + per-lane top-k update for one tile.
    auto computeTile = [&](Frag (&a)[4], int tile) {
        const int m0 = mbase + tile * 16;
        v8f acc = {0.f, 0.f, 0.f, 0.f, 0.f, 0.f, 0.f, 0.f};
#pragma unroll
        for (int kc = 0; kc < 4; ++kc)
            acc = __builtin_amdgcn_wmma_f32_16x16x32_f16(
                    false, a[kc].h, false, bq[kc].h, (short)0, acc, false, false);

        // acc[r] = score(mem row m0 + r + 8*lhi, query qtile + lmod)
#pragma unroll
        for (int r = 0; r < 8; ++r) {
            float s  = acc[r];
            int   id = m0 + r + (lhi ? 8 : 0);
            if (s > tv[0]) {               // rare after warmup; exec-masked
#pragma unroll
                for (int i = 0; i < TOPK - 1; ++i) {
                    bool  cn = s > tv[i + 1];
                    float nv = cn ? tv[i + 1] : s;
                    int   ni = cn ? ti[i + 1] : id;
                    bool  c0 = (i == 0) ? true : (s > tv[i]);
                    tv[i] = c0 ? nv : tv[i];
                    ti[i] = c0 ? ni : ti[i];
                }
                if (s > tv[TOPK - 1]) { tv[TOPK - 1] = s; ti[TOPK - 1] = id; }
            }
        }
    };

    // --- software-pipelined main loop: 2 fragment sets, unrolled by 2 ---
    Frag a0[4], a1[4];
    loadTile(a0, 0);
    for (int tile = 0; tile < MTILES; tile += 2) {
        __builtin_prefetch(kbase + (size_t)(tile + 4) * (16 * CK), 0, 1); // L2 stream-ahead
        loadTile(a1, tile + 1);          // issue loads for tile+1 ...
        computeTile(a0, tile);           // ... while computing tile
        if (tile + 2 < MTILES) loadTile(a0, tile + 2);
        computeTile(a1, tile + 1);
    }

    // --- merge lane L with lane L^16 (same query, disjoint rows):
    // top-32 of two ascending 32-lists = elementwise max(a[i], b[31-i]) ---
#pragma unroll
    for (int i = 0; i < TOPK; ++i) {
        float ov = __shfl_xor(tv[TOPK - 1 - i], 16);
        int   oi = __shfl_xor(ti[TOPK - 1 - i], 16);
        bool  own = tv[i] > ov;
        float mv = own ? tv[i] : ov;
        int   mi = own ? ti[i] : oi;
        if (lhi == 0) {
            cvs[lmod][wv * 32 + i] = mv;
            cis[lmod][wv * 32 + i] = mi;
        }
    }
    __syncthreads();

    // --- exact top-32 over 256 candidates: 16 lanes per query, extract-max ---
    {
        const int q  = t >> 4;   // query 0..15 (16 consecutive lanes = one group)
        const int sl = t & 15;
        for (int r = 0; r < TOPK; ++r) {
            float bv = NEG_BIG;
            int   bp = sl * 16;
            for (int j = 0; j < 16; ++j) {
                float v = cvs[q][sl * 16 + j];
                bool  g = v > bv;
                bp = g ? (sl * 16 + j) : bp;
                bv = g ? v : bv;
            }
#pragma unroll
            for (int off = 1; off < 16; off <<= 1) {
                float ov = __shfl_xor(bv, off);
                int   op = __shfl_xor(bp, off);
                bool  take = (ov > bv) || (ov == bv && op < bp);
                bv = take ? ov : bv;
                bp = take ? op : bp;
            }
            if (sl == 0) { selv[q][r] = bv; seli[q][r] = cis[q][bp]; }
            if ((bp >> 4) == sl) cvs[q][bp] = NEG_BIG;   // owner clears
        }

        // --- softmax over the 32 selected scores (selv[q][0] is the max) ---
        float vmax = selv[q][0];
        float e0 = __expf((selv[q][sl]      - vmax) * TAUINV);
        float e1 = __expf((selv[q][sl + 16] - vmax) * TAUINV);
        float ps = e0 + e1;
#pragma unroll
        for (int off = 1; off < 16; off <<= 1) ps += __shfl_xor(ps, off);
        float rinv = 1.0f / ps;
        attnw[q][sl]      = e0 * rinv;
        attnw[q][sl + 16] = e1 * rinv;
    }
    __syncthreads();

    // --- weighted gather of V columns: thread = (channel group, query) ---
    {
        const int qg = t & 15;
        const int cg = t >> 4;
        const int qglob = qtile + qg;
        float w[TOPK];
        int   id[TOPK];
#pragma unroll
        for (int k = 0; k < TOPK; ++k) { w[k] = attnw[qg][k]; id[k] = seli[qg][k]; }
        for (int c = cg; c < CV; c += 16) {
            const float* vrow = V + (size_t)c * NM;
            float a = 0.0f;
#pragma unroll
            for (int k = 0; k < TOPK; ++k) a += w[k] * vrow[id[k]];
            out[(size_t)c * NQ + qglob] = a;   // coalesced over qg
        }
    }
}

// ---------------------------------------------------------------------------
extern "C" void kernel_launch(void* const* d_in, const int* in_sizes, int n_in,
                              void* d_out, int out_size, void* d_ws, size_t ws_size,
                              hipStream_t stream) {
    const float* Q = (const float*)d_in[0];   // (1,128,64,64)
    const float* K = (const float*)d_in[1];   // (1,128,640,64)
    const float* V = (const float*)d_in[2];   // (1,257,640,64)
    float* out = (float*)d_out;               // (1,257,64,64)

    _Float16* Qh = (_Float16*)d_ws;           // [NQ][CK]  1 MB
    _Float16* Kh = Qh + (size_t)NQ * CK;      // [NM][CK] 10 MB

    norm_to_f16<CK><<<NQ / 256, 256, 0, stream>>>(Q, Qh, NQ);
    norm_to_f16<CK><<<NM / 256, 256, 0, stream>>>(K, Kh, NM);
    attn_topk_kernel<<<NQ / 16, 256, 0, stream>>>(Qh, Kh, V, out);
}
